// LangRelContextBlock_71064528879970
// MI455X (gfx1250) — compile-verified, matrix operand order checked by reading, not run on hardware
//
#include <hip/hip_runtime.h>
#include <hip/hip_bf16.h>
#include <math.h>

// LangRelContextBlock for MI455X (gfx1250, wave32, WMMA f16 16x16x32).
// B=64, N=256, H=256, K=16. Dense math via v_wmma_f32_16x16x32_f16;
// neighbor-feature gather staged via global_load_async_to_lds_b128 (ASYNCcnt).

#define H    256
#define NPTS 256
#define BATCH 64
#define KNN  16

typedef __attribute__((ext_vector_type(16))) _Float16 v16h;
typedef __attribute__((ext_vector_type(8)))  _Float16 v8h;
typedef __attribute__((ext_vector_type(8)))  float    v8f;

__device__ __forceinline__ v8f wmma16(v16h a, v16h b, v8f c) {
  // (neg_a, A, neg_b, B, c_mod, C, reuse_a, reuse_b)
  return __builtin_amdgcn_wmma_f32_16x16x32_f16(false, a, false, b, (short)0, c,
                                                false, false);
}

// Load a 16x32 f16 A-fragment from a row-major LDS tile.
// ISA layout: lane l holds row (l&15); K values {kb..kb+7} and {kb+16..kb+23},
// kb = 8*(l>>4). Two 16-byte LDS loads per lane.
__device__ __forceinline__ v16h load_a_lds(const _Float16* base, int stride,
                                           int kt, int lane) {
  const _Float16* p = base + (size_t)(lane & 15) * stride + kt * 32 + ((lane >> 4) << 3);
  v8h lo = *(const v8h*)(p);
  v8h hi = *(const v8h*)(p + 16);
  return __builtin_shufflevector(lo, hi, 0,1,2,3,4,5,6,7,8,9,10,11,12,13,14,15);
}

// ---------------------------------------------------------------------------
// Pack a row-major f32 weight W (K x N) into WMMA B-fragment layout:
// tiles of 32x16, tile index = nt*(K/32)+kt (kt contiguous for streaming),
// lane l holds col nt*16+(l&15), K bands {kb..kb+7, kb+16..kb+23}, kb=8*(l>>4),
// 16 f16 contiguous per lane -> one 32-byte load in the GEMM.
// ---------------------------------------------------------------------------
__global__ void pack_b_kernel(const float* __restrict__ W,
                              _Float16* __restrict__ out, int K, int N) {
  int gid = blockIdx.x * blockDim.x + threadIdx.x;
  int ktiles = K >> 5;
  int total = ktiles * (N >> 4) * 32;
  if (gid >= total) return;
  int lane = gid & 31;
  int tile = gid >> 5;
  int nt = tile / ktiles;
  int kt = tile - nt * ktiles;
  int n  = nt * 16 + (lane & 15);
  int kb = kt * 32 + ((lane >> 4) << 3);
  _Float16* dst = out + (size_t)tile * 512 + lane * 16;
#pragma unroll
  for (int j = 0; j < 8; ++j) {
    dst[j]     = (_Float16)W[(size_t)(kb + j)      * N + n];
    dst[j + 8] = (_Float16)W[(size_t)(kb + 16 + j) * N + n];
  }
}

// ---------------------------------------------------------------------------
// kNN: one block per (b,n). Packed (dist_bits<<32)|idx keys give top_k's
// lowest-index tie-break. Emits idx + float4(rel.xyz, log1p(dist+1e-6)).
// ---------------------------------------------------------------------------
__global__ __launch_bounds__(256) void knn_kernel(const float* __restrict__ centers,
                                                  int* __restrict__ knn,
                                                  float4* __restrict__ rel4) {
  __shared__ float sC[NPTS][3];
  __shared__ unsigned long long sKey[NPTS];
  __shared__ unsigned long long sRed[NPTS];
  const int tid = threadIdx.x;
  const int b = blockIdx.x >> 8, n = blockIdx.x & (NPTS - 1);
  sC[tid][0] = centers[((size_t)(b * NPTS) + tid) * 3 + 0];
  sC[tid][1] = centers[((size_t)(b * NPTS) + tid) * 3 + 1];
  sC[tid][2] = centers[((size_t)(b * NPTS) + tid) * 3 + 2];
  __syncthreads();
  const float cx = sC[n][0], cy = sC[n][1], cz = sC[n][2];
  float dx = sC[tid][0] - cx, dy = sC[tid][1] - cy, dz = sC[tid][2] - cz;
  float dist = sqrtf(dx * dx + dy * dy + dz * dz);
  sKey[tid] = ((unsigned long long)__float_as_uint(dist) << 32) | (unsigned)tid;
  __syncthreads();
  for (int round = 0; round < KNN; ++round) {
    sRed[tid] = sKey[tid];
    __syncthreads();
    for (int s = NPTS / 2; s > 0; s >>= 1) {
      if (tid < s) {
        unsigned long long o = sRed[tid + s];
        if (o < sRed[tid]) sRed[tid] = o;
      }
      __syncthreads();
    }
    unsigned long long win = sRed[0];
    int widx = (int)(unsigned)(win & 0xffffffffull);
    if (tid == widx) sKey[tid] = ~0ull;
    if (tid == 0) {
      float wx = sC[widx][0] - cx, wy = sC[widx][1] - cy, wz = sC[widx][2] - cz;
      float d = __uint_as_float((unsigned)(win >> 32)) + 1e-6f;
      knn[(size_t)blockIdx.x * KNN + round] = widx;
      rel4[(size_t)blockIdx.x * KNN + round] = make_float4(wx, wy, wz, log1pf(d));
    }
    __syncthreads();
  }
}

// ---------------------------------------------------------------------------
// Per-batch text gate / bias (tiny).
// ---------------------------------------------------------------------------
__global__ __launch_bounds__(256) void text_kernel(
    const float* __restrict__ text, const float* __restrict__ w_gate,
    const float* __restrict__ b_gate, const float* __restrict__ w_tbias,
    const float* __restrict__ b_tbias, float* __restrict__ gatev,
    float* __restrict__ tbiasv) {
  __shared__ float sT[H];
  const int b = blockIdx.x, h = threadIdx.x;
  sT[h] = text[(size_t)b * H + h];
  __syncthreads();
  float ag = b_gate[h], ab = b_tbias[h];
  for (int k = 0; k < H; ++k) {
    float tv = sT[k];
    ag += tv * w_gate[(size_t)k * H + h];
    ab += tv * w_tbias[(size_t)k * H + h];
  }
  gatev[(size_t)b * H + h]  = 1.f / (1.f + expf(-ag));
  tbiasv[(size_t)b * H + h] = ab;
}

// ---------------------------------------------------------------------------
// msgv = ReLU(feat @ w_msg + b_msg): (B*N,256)x(256,256) WMMA GEMM.
// msg depends only on the neighbor node -> computed once per node (16x fewer
// FLOPs than per-edge). A-fragments hoisted to registers, reused for 4 tiles.
// ---------------------------------------------------------------------------
__global__ __launch_bounds__(128) void msg_kernel(const float* __restrict__ feat,
                                                  const _Float16* __restrict__ wmp,
                                                  const float* __restrict__ b_msg,
                                                  float* __restrict__ msgv) {
  __shared__ _Float16 sA[16][H];
  const int tid = threadIdx.x;
  const size_t row0 = (size_t)blockIdx.x * 16;
  for (int i = tid; i < 16 * H; i += 128)
    sA[i >> 8][i & (H - 1)] = (_Float16)feat[row0 * H + i];
  __syncthreads();
  const int wave = tid >> 5, lane = tid & 31;
  const int colo = lane & 15, rbase = (lane >> 4) << 3;
  v16h ag[8];
#pragma unroll
  for (int kt = 0; kt < 8; ++kt) ag[kt] = load_a_lds(&sA[0][0], H, kt, lane);
  for (int t = 0; t < 4; ++t) {
    int nt = wave * 4 + t;
    v8f acc = {};
#pragma unroll
    for (int kt = 0; kt < 8; ++kt) {
      v16h bm = *(const v16h*)(wmp + ((size_t)(nt * 8 + kt)) * 512 + lane * 16);
      acc = wmma16(ag[kt], bm, acc);
    }
    int h = nt * 16 + colo;
#pragma unroll
    for (int v = 0; v < 8; ++v)
      msgv[(row0 + rbase + v) * H + h] = fmaxf(acc[v] + b_msg[h], 0.f);
  }
}

// ---------------------------------------------------------------------------
// Fused per-node kernel. One block per (b,n); M=16 = the node's neighbors.
//   neighbor feats staged with global_load_async_to_lds_b128 (ASYNCcnt),
//   geom MLP layer1 (4->256, VALU) overlapped with the async DMA,
//   geom_emb GEMM (WMMA) + gate/bias/tanh epilogue, edge1 GEMM (WMMA),
//   edge logits + softmax, ctx gather.
// ---------------------------------------------------------------------------
__global__ __launch_bounds__(128) void fused_node_kernel(
    const float* __restrict__ feat, const int* __restrict__ knn,
    const float4* __restrict__ rel4, const float* __restrict__ gatev,
    const float* __restrict__ tbiasv, const float* __restrict__ w_geom1,
    const float* __restrict__ b_geom1, const _Float16* __restrict__ w2p,
    const float* __restrict__ b_geom2, const _Float16* __restrict__ we1p,
    const float* __restrict__ b_edge1, const float* __restrict__ w_edge2,
    const float* __restrict__ b_edge2, const float* __restrict__ msgv,
    float* __restrict__ ctx) {
  __shared__ float    sFeatN[KNN][H];   // neighbor features (async DMA target)
  __shared__ float    sFeatJ[H];        // this node's feature
  __shared__ _Float16 sG1[KNN][H];      // relu(geom @ w_geom1) (A for GEMM 1)
  __shared__ _Float16 sEdge[KNN][H];    // tanh edge features  (A for GEMM 2)
  __shared__ float    sE1[KNN][H / 2];  // relu(edge @ w_edge1)
  __shared__ float    sGeom[KNN][4];
  __shared__ int      sIdx[KNN];
  __shared__ float    sLog[KNN];
  __shared__ float    sAlpha[KNN];
  __shared__ float    sRed[KNN][8];

  const int tid = threadIdx.x;
  const int node = blockIdx.x;
  const int b = node >> 8;

  if (tid < KNN) sIdx[tid] = knn[(size_t)node * KNN + tid];
  if (tid < KNN * 4) {
    int r = tid >> 2, c = tid & 3;
    float4 g = rel4[(size_t)node * KNN + r];
    sGeom[r][c] = (c == 0) ? g.x : (c == 1) ? g.y : (c == 2) ? g.z : g.w;
  }
  __syncthreads();

  // Async global->LDS DMA of the 16x256 f32 neighbor-feature gather.
  // Per lane: 16 bytes global -> LDS, no VGPR data; tracked by ASYNCcnt.
  {
    unsigned ldsbase = (unsigned)(size_t)(&sFeatN[0][0]);
#pragma unroll
    for (int i = 0; i < 8; ++i) {
      int e = i * 512 + tid * 4;  // float index into the 16x256 tile
      int r = e >> 8, h = e & (H - 1);
      const float* gp = feat + ((size_t)(b * NPTS) + sIdx[r]) * H + h;
      unsigned dst = ldsbase + (unsigned)e * 4u;
      asm volatile("global_load_async_to_lds_b128 %0, %1, off"
                   :: "v"(dst), "v"(gp)
                   : "memory");
    }
  }

  // Overlap with the DMA: node feature + tiny geom layer 1 (16x4 @ 4x256).
  for (int i = tid; i < H; i += 128) sFeatJ[i] = feat[(size_t)node * H + i];
  for (int h = tid; h < H; h += 128) {
    float w0 = w_geom1[h], w1 = w_geom1[H + h];
    float w2 = w_geom1[2 * H + h], w3 = w_geom1[3 * H + h];
    float bb = b_geom1[h];
    for (int r = 0; r < KNN; ++r) {
      float g = bb + sGeom[r][0] * w0 + sGeom[r][1] * w1 + sGeom[r][2] * w2 +
                sGeom[r][3] * w3;
      sG1[r][h] = (_Float16)fmaxf(g, 0.f);
    }
  }
  asm volatile("s_wait_asynccnt 0x0" ::: "memory");
  __syncthreads();

  const int wave = tid >> 5, lane = tid & 31;
  const int colo = lane & 15, rbase = (lane >> 4) << 3;

  // geom_emb = relu(sG1 @ w_geom2 + b2) -> *gate+bias -> tanh(+feats) -> sEdge
  {
    v16h ag[8];
#pragma unroll
    for (int kt = 0; kt < 8; ++kt) ag[kt] = load_a_lds(&sG1[0][0], H, kt, lane);
    for (int t = 0; t < 4; ++t) {
      int nt = wave * 4 + t;
      v8f acc = {};
#pragma unroll
      for (int kt = 0; kt < 8; ++kt) {
        v16h bm = *(const v16h*)(w2p + ((size_t)(nt * 8 + kt)) * 512 + lane * 16);
        acc = wmma16(ag[kt], bm, acc);
      }
      int h = nt * 16 + colo;
      float gv = gatev[b * H + h], tb = tbiasv[b * H + h];
      float b2 = b_geom2[h], fj = sFeatJ[h];
#pragma unroll
      for (int v = 0; v < 8; ++v) {
        int r = rbase + v;
        float ge = fmaxf(acc[v] + b2, 0.f);
        float ef = tanhf(fj + sFeatN[r][h] + ge * gv + tb);
        sEdge[r][h] = (_Float16)ef;
      }
    }
  }
  __syncthreads();

  // e1 = relu(sEdge @ w_edge1 + b_e1): 16x256 @ 256x128
  {
    v16h ag[8];
#pragma unroll
    for (int kt = 0; kt < 8; ++kt) ag[kt] = load_a_lds(&sEdge[0][0], H, kt, lane);
    for (int t = 0; t < 2; ++t) {
      int nt = wave * 2 + t;
      v8f acc = {};
#pragma unroll
      for (int kt = 0; kt < 8; ++kt) {
        v16h bm = *(const v16h*)(we1p + ((size_t)(nt * 8 + kt)) * 512 + lane * 16);
        acc = wmma16(ag[kt], bm, acc);
      }
      int h = nt * 16 + colo;
#pragma unroll
      for (int v = 0; v < 8; ++v)
        sE1[rbase + v][h] = fmaxf(acc[v] + b_edge1[h], 0.f);
    }
  }
  __syncthreads();

  // logits (length-128 dot) + softmax over the 16 neighbors
  {
    int r = tid >> 3, s = (tid & 7) * 16;
    float p = 0.f;
    for (int j = 0; j < 16; ++j) p += sE1[r][s + j] * w_edge2[s + j];
    sRed[r][tid & 7] = p;
  }
  __syncthreads();
  if (tid < KNN) {
    float l = b_edge2[0];
    for (int j = 0; j < 8; ++j) l += sRed[tid][j];
    sLog[tid] = l;
  }
  __syncthreads();
  if (tid < KNN) {
    float m = -1e30f;
    for (int j = 0; j < KNN; ++j) m = fmaxf(m, sLog[j]);
    float sum = 0.f;
    for (int j = 0; j < KNN; ++j) sum += expf(sLog[j] - m);
    sAlpha[tid] = expf(sLog[tid] - m) / sum;
  }
  __syncthreads();

  // ctx = sum_r alpha_r * msgv[neighbor_r]
  for (int h = tid; h < H; h += 128) {
    float c = 0.f;
    for (int r = 0; r < KNN; ++r)
      c += sAlpha[r] * msgv[((size_t)(b * NPTS) + sIdx[r]) * H + h];
    ctx[(size_t)node * H + h] = c;
  }
}

// ---------------------------------------------------------------------------
// out = relu([feat,ctx] @ w_out + b_out); x = feat + out; LayerNorm(x).
// (B*N,512)x(512,256) WMMA GEMM with fused residual + LN epilogue.
// K split into two halves so only 8 A-fragments are live at a time.
// ---------------------------------------------------------------------------
__global__ __launch_bounds__(128) void out_ln_kernel(
    const float* __restrict__ feat, const float* __restrict__ ctx,
    const _Float16* __restrict__ wop, const float* __restrict__ b_out,
    const float* __restrict__ gamma, const float* __restrict__ beta,
    float* __restrict__ out) {
  __shared__ _Float16 sA[16][2 * H];
  __shared__ float    sX[16][H];
  __shared__ float    sS[16][8], sS2[16][8];
  __shared__ float    sMu[16], sRv[16];
  const int tid = threadIdx.x;
  const int row0 = blockIdx.x * 16;
  for (int i = tid; i < 16 * H; i += 128) {
    int r = i >> 8, h = i & (H - 1);
    sA[r][h]     = (_Float16)feat[(size_t)(row0 + r) * H + h];
    sA[r][H + h] = (_Float16)ctx[(size_t)(row0 + r) * H + h];
  }
  __syncthreads();
  const int wave = tid >> 5, lane = tid & 31;
  const int colo = lane & 15, rbase = (lane >> 4) << 3;
  v8f acc[4];
#pragma unroll
  for (int t = 0; t < 4; ++t) acc[t] = (v8f){};
  for (int kh = 0; kh < 2; ++kh) {
    v16h ag[8];
#pragma unroll
    for (int kk = 0; kk < 8; ++kk)
      ag[kk] = load_a_lds(&sA[0][0], 2 * H, kh * 8 + kk, lane);
    for (int t = 0; t < 4; ++t) {
      int nt = wave * 4 + t;
#pragma unroll
      for (int kk = 0; kk < 8; ++kk) {
        v16h bm = *(const v16h*)(wop + ((size_t)(nt * 16 + kh * 8 + kk)) * 512 +
                                 lane * 16);
        acc[t] = wmma16(ag[kk], bm, acc[t]);
      }
    }
  }
  for (int t = 0; t < 4; ++t) {
    int nt = wave * 4 + t;
    int h = nt * 16 + colo;
#pragma unroll
    for (int v = 0; v < 8; ++v) {
      int r = rbase + v;
      float o = fmaxf(acc[t][v] + b_out[h], 0.f);
      sX[r][h] = feat[(size_t)(row0 + r) * H + h] + o;
    }
  }
  __syncthreads();
  {
    int r = tid >> 3, s = (tid & 7) * 32;
    float su = 0.f, sq = 0.f;
    for (int j = 0; j < 32; ++j) {
      float x = sX[r][s + j];
      su += x; sq += x * x;
    }
    sS[r][tid & 7] = su; sS2[r][tid & 7] = sq;
  }
  __syncthreads();
  if (tid < 16) {
    float su = 0.f, sq = 0.f;
    for (int j = 0; j < 8; ++j) { su += sS[tid][j]; sq += sS2[tid][j]; }
    float mu = su / (float)H;
    float var = sq / (float)H - mu * mu;
    sMu[tid] = mu;
    sRv[tid] = rsqrtf(var + 1e-5f);
  }
  __syncthreads();
  for (int i = tid; i < 16 * H; i += 128) {
    int r = i >> 8, h = i & (H - 1);
    out[(size_t)(row0 + r) * H + h] =
        (sX[r][h] - sMu[r]) * sRv[r] * gamma[h] + beta[h];
  }
}

// ---------------------------------------------------------------------------
extern "C" void kernel_launch(void* const* d_in, const int* in_sizes, int n_in,
                              void* d_out, int out_size, void* d_ws,
                              size_t ws_size, hipStream_t stream) {
  (void)in_sizes; (void)n_in; (void)out_size; (void)ws_size;
  const float* feat    = (const float*)d_in[0];
  const float* centers = (const float*)d_in[1];
  const float* text    = (const float*)d_in[2];
  const float* w_geom1 = (const float*)d_in[3];
  const float* b_geom1 = (const float*)d_in[4];
  const float* w_geom2 = (const float*)d_in[5];
  const float* b_geom2 = (const float*)d_in[6];
  const float* w_gate  = (const float*)d_in[7];
  const float* b_gate  = (const float*)d_in[8];
  const float* w_tbias = (const float*)d_in[9];
  const float* b_tbias = (const float*)d_in[10];
  const float* w_edge1 = (const float*)d_in[11];
  const float* b_edge1 = (const float*)d_in[12];
  const float* w_edge2 = (const float*)d_in[13];
  const float* b_edge2 = (const float*)d_in[14];
  const float* w_msg   = (const float*)d_in[15];
  const float* b_msg   = (const float*)d_in[16];
  const float* w_out   = (const float*)d_in[17];
  const float* b_out   = (const float*)d_in[18];
  const float* gamma   = (const float*)d_in[19];
  const float* beta    = (const float*)d_in[20];
  float* out = (float*)d_out;

  char* ws = (char*)d_ws;
  auto bump = [&](size_t bytes) -> void* {
    void* p = (void*)ws;
    ws += (bytes + 255) & ~(size_t)255;
    return p;
  };
  const int rows = BATCH * NPTS;  // 16384
  int*      knn_idx = (int*)bump((size_t)rows * KNN * sizeof(int));
  float4*   rel4    = (float4*)bump((size_t)rows * KNN * sizeof(float4));
  float*    gatev   = (float*)bump((size_t)BATCH * H * sizeof(float));
  float*    tbiasv  = (float*)bump((size_t)BATCH * H * sizeof(float));
  _Float16* w2p     = (_Float16*)bump((size_t)H * H * sizeof(_Float16));
  _Float16* we1p    = (_Float16*)bump((size_t)H * (H / 2) * sizeof(_Float16));
  _Float16* wmp     = (_Float16*)bump((size_t)H * H * sizeof(_Float16));
  _Float16* wop     = (_Float16*)bump((size_t)2 * H * H * sizeof(_Float16));
  float*    msgv    = (float*)bump((size_t)rows * H * sizeof(float));
  float*    ctxv    = (float*)bump((size_t)rows * H * sizeof(float));

  // Pack GEMM weights to the WMMA B-fragment f16 layout.
  pack_b_kernel<<<(H / 32) * (H / 16) * 32 / 256, 256, 0, stream>>>(w_geom2, w2p, H, H);
  pack_b_kernel<<<(H / 32) * ((H / 2) / 16) * 32 / 256, 256, 0, stream>>>(w_edge1, we1p, H, H / 2);
  pack_b_kernel<<<(H / 32) * (H / 16) * 32 / 256, 256, 0, stream>>>(w_msg, wmp, H, H);
  pack_b_kernel<<<((2 * H) / 32) * (H / 16) * 32 / 256, 256, 0, stream>>>(w_out, wop, 2 * H, H);

  knn_kernel<<<rows, NPTS, 0, stream>>>(centers, knn_idx, rel4);
  text_kernel<<<BATCH, H, 0, stream>>>(text, w_gate, b_gate, w_tbias, b_tbias,
                                       gatev, tbiasv);
  msg_kernel<<<rows / 16, 128, 0, stream>>>(feat, wmp, b_msg, msgv);
  fused_node_kernel<<<rows, 128, 0, stream>>>(feat, knn_idx, rel4, gatev, tbiasv,
                                              w_geom1, b_geom1, w2p, b_geom2,
                                              we1p, b_edge1, w_edge2, b_edge2,
                                              msgv, ctxv);
  out_ln_kernel<<<rows / 16, 128, 0, stream>>>(feat, ctxv, wop, b_out, gamma,
                                               beta, out);
}
